// LCNet_57595511439609
// MI455X (gfx1250) — matrix operand order repaired and checked by reading, not compiled, not run on hardware
//
#include <hip/hip_runtime.h>
#include <hip/hip_fp16.h>

typedef __attribute__((ext_vector_type(16))) _Float16 v16h;
typedef __attribute__((ext_vector_type(8)))  float    v8f;

#define TPB 256

// ---------------- degree kernels ----------------

__global__ void k_fill(float* __restrict__ p, float v, int n) {
  int i = blockIdx.x * blockDim.x + threadIdx.x;
  if (i < n) p[i] = v;
}

__global__ void k_degcount(const long long* __restrict__ dst, float* __restrict__ deg, int E) {
  int e = blockIdx.x * blockDim.x + threadIdx.x;
  if (e < E) atomicAdd(&deg[(int)dst[e]], 1.0f);
}

__global__ void k_rsqrt(float* __restrict__ p, int n) {
  int i = blockIdx.x * blockDim.x + threadIdx.x;
  if (i < n) p[i] = rsqrtf(p[i]);   // deg >= 1 always (self-loop)
}

__global__ void k_f32_to_f16(const float* __restrict__ in, _Float16* __restrict__ out, int n) {
  int i = blockIdx.x * blockDim.x + threadIdx.x;
  if (i < n) out[i] = (_Float16)in[i];
}

// ---------------- W repack: f32 [K][Nc] -> f16 WMMA B-fragment order ----------------
// Bp linear index = ((ct*NS + s)*32 + lane)*16 + t
//   lane 0-15 : K = s*32 + [0..15],  lane 16-31 : K = s*32 + [16..31]  (t = K offset)
//   column    = ct*16 + (lane & 15)
__global__ void k_pack_w(const float* __restrict__ W, _Float16* __restrict__ Bp, int K, int Nc) {
  int i = blockIdx.x * blockDim.x + threadIdx.x;
  if (i >= K * Nc) return;
  int t    = i & 15;
  int lane = (i >> 4) & 31;
  int rest = i >> 9;            // ct*NS + s
  int NS   = K >> 5;
  int s    = rest % NS;
  int ct   = rest / NS;
  int kb   = s * 32 + (lane >> 4) * 16 + t;
  int ncol = ct * 16 + (lane & 15);
  Bp[i] = (_Float16)W[kb * Nc + ncol];
}

// ---------------- fused WMMA GEMM + bias/self-loop epilogue ----------------
// One wave per 16-row strip: A fragments loaded once, reused across all column tiles.
// Emits H = A@B (for edge gather) and O = b + dinv^2 * H (aggregation init).
template<int K, int NC>
__global__ __launch_bounds__(TPB)
void k_gemm_wmma(const _Float16* __restrict__ A, const _Float16* __restrict__ Bp,
                 const float* __restrict__ bias, const float* __restrict__ dinv,
                 float* __restrict__ H, float* __restrict__ O, int n) {
  constexpr int NS = K / 32;    // k-steps
  constexpr int CT = NC / 16;   // column tiles
  const int lane = threadIdx.x & 31;
  const int widx = blockIdx.x * (blockDim.x >> 5) + (threadIdx.x >> 5);
  if (widx >= (n >> 4)) return;          // wave-uniform: EXEC all-ones for WMMA waves

  const int rt = widx;
  const int hi = lane >> 4;              // half-wave selects K sub-range
  const int ln = lane & 15;
  const int m  = rt * 16 + ln;           // A row handled by this lane

  // A fragments (contiguous k order, same as packed B): 16 contiguous halves per lane.
  // lanes 0-15: K = s*32+[0..15]; lanes 16-31: K = s*32+[16..31]
  v16h af[NS];
  #pragma unroll
  for (int s = 0; s < NS; ++s) {
    union { v16h v; uint4 u[2]; } t;
    const _Float16* arow = A + (size_t)m * K + s * 32 + hi * 16;
    t.u[0] = *(const uint4*)(arow);
    t.u[1] = *(const uint4*)(arow + 8);
    af[s] = t.v;
  }

  #pragma unroll
  for (int ct = 0; ct < CT; ++ct) {
    v8f acc = {};
    #pragma unroll
    for (int s = 0; s < NS; ++s) {
      // pre-packed B fragment: 16 contiguous halves per lane -> two b128 loads
      union { v16h v; uint4 u[2]; } t;
      const _Float16* bp = Bp + ((size_t)(ct * NS + s) * 32 + lane) * 16;
      t.u[0] = *(const uint4*)(bp);
      t.u[1] = *(const uint4*)(bp + 8);
      acc = __builtin_amdgcn_wmma_f32_16x16x32_f16(
          false, af[s], false, t.v, (short)0, acc, false, false);
    }
    // C/D layout: VGPR v -> row rt*16 + hi*8 + v, column = ct*16 + ln
    const int ncol = ct * 16 + ln;
    const float bc = bias[ncol];
    #pragma unroll
    for (int v = 0; v < 8; ++v) {
      const int row = rt * 16 + hi * 8 + v;
      const float hval = acc[v];
      const float di = dinv[row];
      H[(size_t)row * NC + ncol] = hval;
      O[(size_t)row * NC + ncol] = bc + di * di * hval;
    }
  }
}

// ---------------- edge scatter: one (edge, channel) per lane ----------------
__global__ void k_edge_agg(const long long* __restrict__ src, const long long* __restrict__ dst,
                           const float* __restrict__ H, const float* __restrict__ dinv,
                           float* __restrict__ O, int E, int logNc) {
  int t = blockIdx.x * blockDim.x + threadIdx.x;
  int e = t >> logNc;
  if (e >= E) return;
  int c = t & ((1 << logNc) - 1);
  int si = (int)src[e];
  int di = (int)dst[e];
  float norm = dinv[si] * dinv[di];
  atomicAdd(O + (((size_t)di << logNc) + c), norm * H[((size_t)si << logNc) + c]);
}

// ---------------- activations ----------------
// CELU(alpha=1) + deterministic hash dropout, writing f16 GEMM input directly
__global__ void k_celu_drop_f16(const float* __restrict__ in, _Float16* __restrict__ out,
                                int n, unsigned salt) {
  int i = blockIdx.x * blockDim.x + threadIdx.x;
  if (i >= n) return;
  float v = in[i];
  float a = v > 0.0f ? v : (__expf(v) - 1.0f);
  unsigned h = (unsigned)i ^ salt;
  h ^= h >> 16; h *= 0x7feb352dU;
  h ^= h >> 15; h *= 0x846ca68bU;
  h ^= h >> 16;
  a = (h & 1u) ? a * 2.0f : 0.0f;
  out[i] = (_Float16)a;
}

// final CELU -> f32 output
__global__ void k_celu_f32(const float* __restrict__ in, float* __restrict__ out, int n) {
  int i = blockIdx.x * blockDim.x + threadIdx.x;
  if (i >= n) return;
  float v = in[i];
  out[i] = v > 0.0f ? v : (__expf(v) - 1.0f);
}

// ---------------- driver ----------------

static inline int cdiv(int a, int b) { return (a + b - 1) / b; }

extern "C" void kernel_launch(void* const* d_in, const int* in_sizes, int n_in,
                              void* d_out, int out_size, void* d_ws, size_t ws_size,
                              hipStream_t stream) {
  const float*     x  = (const float*)d_in[0];
  const long long* ei = (const long long*)d_in[1];     // int64 edge_index [2][E]
  const float*     Wt[3] = { (const float*)d_in[2], (const float*)d_in[4], (const float*)d_in[6] };
  const float*     bt[3] = { (const float*)d_in[3], (const float*)d_in[5], (const float*)d_in[7] };

  const int n = in_sizes[0] / 64;      // 100000
  const int E = in_sizes[1] / 2;       // 3200000
  const long long* srcI = ei;
  const long long* dstI = ei + E;

  // workspace carve-out (256B aligned)
  char* w = (char*)d_ws;
  auto take = [&](size_t bytes) -> char* {
    char* p = w; w += (bytes + 255) & ~(size_t)255; return p;
  };
  float*    dinv = (float*)take((size_t)n * 4);
  float*    H    = (float*)take((size_t)n * 64 * 4);
  float*    O    = (float*)take((size_t)n * 64 * 4);
  _Float16* Xh   = (_Float16*)take((size_t)n * 64 * 2);
  _Float16* Bp   = (_Float16*)take(64 * 64 * 2);

  // deg = 1 (self loop) + in-degree; dinv = rsqrt(deg)
  k_fill<<<cdiv(n, TPB), TPB, 0, stream>>>(dinv, 1.0f, n);
  k_degcount<<<cdiv(E, TPB), TPB, 0, stream>>>(dstI, dinv, E);
  k_rsqrt<<<cdiv(n, TPB), TPB, 0, stream>>>(dinv, n);

  // initial f16 staging of x
  k_f32_to_f16<<<cdiv(n * 64, TPB), TPB, 0, stream>>>(x, Xh, n * 64);

  const int KL[3]       = { 64, 32, 64 };
  const int NL[3]       = { 32, 64, 32 };
  const unsigned salt[3] = { 0x9E3779B1u, 0x85EBCA77u, 0u };

  const int gemmBlocks = cdiv(n / 16, TPB / 32);

  for (int l = 0; l < 3; ++l) {
    const int K = KL[l], Nc = NL[l];
    const int logNc = (Nc == 32) ? 5 : 6;

    // pack W into WMMA B-fragment order (f16)
    k_pack_w<<<cdiv(K * Nc, TPB), TPB, 0, stream>>>(Wt[l], Bp, K, Nc);

    // H = Xh @ W; O = b + dinv^2 * H   (fused epilogue)
    if (l == 1) {
      k_gemm_wmma<32, 64><<<gemmBlocks, TPB, 0, stream>>>(Xh, Bp, bt[l], dinv, H, O, n);
    } else {
      k_gemm_wmma<64, 32><<<gemmBlocks, TPB, 0, stream>>>(Xh, Bp, bt[l], dinv, H, O, n);
    }

    // scatter-add edge messages (L2-resident gather + f32 atomics)
    const int total = E * Nc;   // <= 204,800,000 fits int
    k_edge_agg<<<cdiv(total, TPB), TPB, 0, stream>>>(srcI, dstI, H, dinv, O, E, logNc);

    if (l < 2) {
      // CELU + dropout, emit next layer's f16 GEMM input directly
      k_celu_drop_f16<<<cdiv(n * Nc, TPB), TPB, 0, stream>>>(O, Xh, n * Nc, salt[l]);
    } else {
      // final CELU -> f32 output
      k_celu_f32<<<cdiv(n * Nc, TPB), TPB, 0, stream>>>(O, (float*)d_out, n * Nc);
    }
  }
}
